// JambaBlock_69217692942439
// MI455X (gfx1250) — compile-verified, matrix operand order checked by reading, not run on hardware
//
#include <hip/hip_runtime.h>
#include <hip/hip_bf16.h>

typedef __bf16 bf16_t;
typedef __attribute__((ext_vector_type(16))) __bf16 v16bf;
typedef __attribute__((ext_vector_type(8)))  float  v8f;

#define B_SZ 2
#define L_SZ 8192
#define D_SZ 1024
#define DI   2048
#define DS   16
#define DTR  64
#define NGRP 96            // DTR + 2*DS
#define BL   (B_SZ * L_SZ) // 16384
#define GCH  64            // scan chunks
#define LCH  128           // chunk length (GCH*LCH == L_SZ)

// ---------------- CDNA5 async global->LDS helpers ----------------
__device__ __forceinline__ void async_ld_b128(unsigned lds_off, const void* gptr) {
  // GLOBAL_LOAD_ASYNC_TO_LDS_B128: per-lane 16B DMA, tracked by ASYNCcnt
  asm volatile("global_load_async_to_lds_b128 %0, %1, off"
               :: "v"(lds_off), "v"((unsigned long long)(size_t)gptr)
               : "memory");
}
__device__ __forceinline__ void wait_async_4() {
  asm volatile("s_wait_asynccnt 0x4" ::: "memory");
}
__device__ __forceinline__ void wait_async_0() {
  asm volatile("s_wait_asynccnt 0x0" ::: "memory");
}

// ---------------- RMSNorm: one row (D=1024) per 256-thread block ----------------
__global__ void k_rmsnorm(const float* __restrict__ x, bf16_t* __restrict__ out) {
  __shared__ float red[256];
  int row = blockIdx.x;
  const float* xr = x + (size_t)row * D_SZ;
  float v[4]; float s = 0.f;
#pragma unroll
  for (int i = 0; i < 4; ++i) { v[i] = xr[threadIdx.x + 256 * i]; s += v[i] * v[i]; }
  red[threadIdx.x] = s;
  __syncthreads();
  for (int off = 128; off > 0; off >>= 1) {
    if ((int)threadIdx.x < off) red[threadIdx.x] += red[threadIdx.x + off];
    __syncthreads();
  }
  float n = sqrtf(red[0]);
  float sc = 32.0f / fmaxf(n, 1e-12f); // * sqrt(D)
  bf16_t* o = out + (size_t)row * D_SZ;
#pragma unroll
  for (int i = 0; i < 4; ++i) o[threadIdx.x + 256 * i] = (bf16_t)(v[i] * sc);
}

// ---------------- Weight transpose + fp32->bf16: out[n*K+k] = in[k*N+n] ----------------
__global__ void k_wT(const float* __restrict__ in, bf16_t* __restrict__ out, int K, int N) {
  int idx = blockIdx.x * 256 + threadIdx.x;
  if (idx >= K * N) return;
  int n = idx / K, k = idx % K;
  out[idx] = (bf16_t)in[(size_t)k * N + n];
}

// ---------------- bf16 WMMA GEMM: C[M,N] = A[M,K] * BT[N,K]^T ----------------
// 128x128 macro-tile, 8 waves, wave tile 32x64 (2x4 WMMA), BK=32,
// double-buffered LDS filled by async global->LDS DMA; last stage peeled.
#define BM 128
#define BN 128
#define BK 32
#define LSTR 24 // LDS row stride in dwords (keeps 16B alignment, spreads banks)

__global__ void __launch_bounds__(256) k_gemm_bf16(
    const bf16_t* __restrict__ A, const bf16_t* __restrict__ BT,
    float* __restrict__ C, int M, int N, int K, int lda, int ldc) {
  __shared__ unsigned int ldsA[2][BM * LSTR];
  __shared__ unsigned int ldsB[2][BN * LSTR];
  int tid = threadIdx.x;
  int lane = tid & 31, w = tid >> 5;
  int wm = (w & 3) * 32;  // wave row base in tile (4 waves along M)
  int wn = (w >> 2) * 64; // wave col base in tile (2 waves along N)
  int tileM = blockIdx.y * BM, tileN = blockIdx.x * BN;

  // per-thread staging coordinates: 512 16B-groups per 128x32 tile, 2 per thread
  int rA0 = tid >> 2, rA1 = (tid + 256) >> 2, kg = tid & 3;
  int rrB0 = tileN + rA0; if (rrB0 >= N) rrB0 = N - 1; // clamp OOB rows (cols store-guarded)
  int rrB1 = tileN + rA1; if (rrB1 >= N) rrB1 = N - 1;

  // loop-invariant LDS dest offsets for both buffers
  unsigned la0[2], la1[2], lb0[2], lb1[2];
#pragma unroll
  for (int buf = 0; buf < 2; ++buf) {
    la0[buf] = (unsigned)(size_t)&ldsA[buf][rA0 * LSTR + kg * 4];
    la1[buf] = (unsigned)(size_t)&ldsA[buf][rA1 * LSTR + kg * 4];
    lb0[buf] = (unsigned)(size_t)&ldsB[buf][rA0 * LSTR + kg * 4];
    lb1[buf] = (unsigned)(size_t)&ldsB[buf][rA1 * LSTR + kg * 4];
  }
  // per-thread global pointers, advanced by BK per stage
  const bf16_t* pa0 = A + (size_t)(tileM + rA0) * lda + kg * 8;
  const bf16_t* pa1 = A + (size_t)(tileM + rA1) * lda + kg * 8;
  const bf16_t* pb0 = BT + (size_t)rrB0 * K + kg * 8;
  const bf16_t* pb1 = BT + (size_t)rrB1 * K + kg * 8;

  v8f acc[2][4] = {};
  int nk = K / BK;
  int o = (lane >> 4) * 4; // dword group per ISA 16-bit A/B fragment layout
  int rsel = lane & 15;

  auto compute = [&](int cur) {
    union Frag { v16bf v; uint4 q[2]; };
    Frag afr[2], bfr[4];
#pragma unroll
    for (int i = 0; i < 2; ++i) {
      const unsigned int* p = &ldsA[cur][(wm + i * 16 + rsel) * LSTR + o];
      afr[i].q[0] = *(const uint4*)p;
      afr[i].q[1] = *(const uint4*)(p + 8);
    }
#pragma unroll
    for (int j = 0; j < 4; ++j) {
      const unsigned int* p = &ldsB[cur][(wn + j * 16 + rsel) * LSTR + o];
      bfr[j].q[0] = *(const uint4*)p;
      bfr[j].q[1] = *(const uint4*)(p + 8);
    }
#pragma unroll
    for (int i = 0; i < 2; ++i)
#pragma unroll
      for (int j = 0; j < 4; ++j)
        acc[i][j] = __builtin_amdgcn_wmma_f32_16x16x32_bf16(
            false, afr[i].v, false, bfr[j].v, (short)0, acc[i][j], false, false);
  };

  // issue stage 0
  async_ld_b128(la0[0], pa0);
  async_ld_b128(la1[0], pa1);
  async_ld_b128(lb0[0], pb0);
  async_ld_b128(lb1[0], pb1);
  pa0 += BK; pa1 += BK; pb0 += BK; pb1 += BK;

  // steady state: always issue next stage, wait for current (<=4 leaves next in flight)
  for (int kk = 0; kk < nk - 1; ++kk) {
    int cur = kk & 1, nxt = cur ^ 1;
    async_ld_b128(la0[nxt], pa0);
    async_ld_b128(la1[nxt], pa1);
    async_ld_b128(lb0[nxt], pb0);
    async_ld_b128(lb1[nxt], pb1);
    pa0 += BK; pa1 += BK; pb0 += BK; pb1 += BK;
    wait_async_4();
    __syncthreads(); // all waves' stage kk visible in LDS
    compute(cur);
    __syncthreads(); // all waves done reading buf `cur` before re-fill
  }
  // peeled last stage
  wait_async_0();
  __syncthreads();
  compute((nk - 1) & 1);

  int rhi = (lane >> 4) * 8;
#pragma unroll
  for (int i = 0; i < 2; ++i)
#pragma unroll
    for (int j = 0; j < 4; ++j) {
      int col = tileN + wn + j * 16 + rsel;
      if (col < N) {
        int row0 = tileM + wm + i * 16 + rhi;
#pragma unroll
        for (int r = 0; r < 8; ++r)
          C[(size_t)(row0 + r) * ldc + col] = acc[i][j][r];
      }
    }
}

// ---------------- depthwise causal conv (k=4) + SiLU, reads x_in half of xr ----------------
__global__ void k_conv_silu(const float* __restrict__ xr, const float* __restrict__ cw,
                            const float* __restrict__ cb, bf16_t* __restrict__ ub) {
  int idx = blockIdx.x * 256 + threadIdx.x; // BL*DI
  int c = idx & (DI - 1);
  int row = idx >> 11;
  int b = row >> 13;
  int t = row & (L_SZ - 1);
  float acc = cb[c];
#pragma unroll
  for (int j = 0; j < 4; ++j) {
    int tt = t - 3 + j;
    if (tt >= 0)
      acc += cw[c * 4 + j] * xr[(size_t)(b * L_SZ + tt) * (2 * DI) + c];
  }
  float sg = 1.f / (1.f + __expf(-acc));
  ub[idx] = (bf16_t)(acc * sg);
}

// ---------------- extract dt cols (0..63) of dbl to bf16 ----------------
__global__ void k_extract_dt(const float* __restrict__ dbl, bf16_t* __restrict__ dtb) {
  int idx = blockIdx.x * 256 + threadIdx.x; // BL*DTR
  int r = idx >> 6, c = idx & 63;
  dtb[idx] = (bf16_t)dbl[(size_t)r * NGRP + c];
}

// ---------------- softplus(x + b_dt[c]) in place ----------------
__global__ void k_softplus_bias(float* __restrict__ delta, const float* __restrict__ bdt) {
  int idx = blockIdx.x * 256 + threadIdx.x; // BL*DI
  int c = idx & (DI - 1);
  float v = delta[idx] + bdt[c];
  delta[idx] = v > 20.f ? v : log1pf(__expf(v));
}

// ---------------- chunked selective scan, pass 1: local scans ----------------
__global__ void k_scan1(const float* __restrict__ delta, const bf16_t* __restrict__ ub,
                        const float* __restrict__ dbl, const float* __restrict__ Alog,
                        float* __restrict__ ypart, float* __restrict__ hF,
                        float* __restrict__ Ssum) {
  int ch = blockIdx.x * 256 + threadIdx.x;
  int b = blockIdx.y, g = blockIdx.z;
  float A[DS], h[DS];
#pragma unroll
  for (int s = 0; s < DS; ++s) { A[s] = -__expf(Alog[ch * DS + s]); h[s] = 0.f; }
  float S = 0.f;
  int t0 = g * LCH;
  for (int t = t0; t < t0 + LCH; ++t) {
    size_t base = (size_t)(b * L_SZ + t);
    float d = delta[base * DI + ch];
    float uu = (float)ub[base * DI + ch];
    float du = d * uu;
    S += d;
    const float* row = dbl + base * NGRP;
    float y = 0.f;
#pragma unroll
    for (int s = 0; s < DS; ++s) {
      float a = __expf(d * A[s]);
      h[s] = a * h[s] + du * row[DTR + s];
      y += h[s] * row[DTR + DS + s];
    }
    ypart[base * DI + ch] = y;
  }
  size_t cidx = (size_t)(b * GCH + g) * DI + ch;
#pragma unroll
  for (int s = 0; s < DS; ++s) hF[cidx * DS + s] = h[s];
  Ssum[cidx] = S;
}

// ---------------- pass 2: sequential combine over 64 chunks ----------------
__global__ void k_scan2(const float* __restrict__ hF, const float* __restrict__ Ssum,
                        const float* __restrict__ Alog, float* __restrict__ hstart) {
  int idx = blockIdx.x * 256 + threadIdx.x; // B*DI = 4096
  int b = idx >> 11, ch = idx & (DI - 1);
  float A[DS], h[DS];
#pragma unroll
  for (int s = 0; s < DS; ++s) { A[s] = -__expf(Alog[ch * DS + s]); h[s] = 0.f; }
  for (int g = 0; g < GCH; ++g) {
    size_t cidx = (size_t)(b * GCH + g) * DI + ch;
#pragma unroll
    for (int s = 0; s < DS; ++s) hstart[cidx * DS + s] = h[s];
    float S = Ssum[cidx];
#pragma unroll
    for (int s = 0; s < DS; ++s) h[s] = __expf(A[s] * S) * h[s] + hF[cidx * DS + s];
  }
}

// ---------------- pass 3: correction + (u*Dp) + SiLU(res) gate, emit bf16 ----------------
__global__ void k_scan3(const float* __restrict__ delta, const bf16_t* __restrict__ ub,
                        const float* __restrict__ dbl, const float* __restrict__ hstart,
                        const float* __restrict__ Alog, const float* __restrict__ Dp,
                        const float* __restrict__ xr, const float* __restrict__ ypart,
                        bf16_t* __restrict__ yb) {
  int ch = blockIdx.x * 256 + threadIdx.x;
  int b = blockIdx.y, g = blockIdx.z;
  float A[DS], hs[DS];
  size_t cidx = (size_t)(b * GCH + g) * DI + ch;
#pragma unroll
  for (int s = 0; s < DS; ++s) {
    A[s] = -__expf(Alog[ch * DS + s]);
    hs[s] = hstart[cidx * DS + s];
  }
  float dp = Dp[ch];
  float cd = 0.f;
  int t0 = g * LCH;
  for (int t = t0; t < t0 + LCH; ++t) {
    size_t base = (size_t)(b * L_SZ + t);
    float d = delta[base * DI + ch];
    cd += d;
    const float* row = dbl + base * NGRP;
    float y = ypart[base * DI + ch];
#pragma unroll
    for (int s = 0; s < DS; ++s)
      y += __expf(A[s] * cd) * hs[s] * row[DTR + DS + s];
    float uu = (float)ub[base * DI + ch];
    y += uu * dp;
    float r = xr[base * (2 * DI) + DI + ch];
    y *= r / (1.f + __expf(-r));
    yb[base * DI + ch] = (bf16_t)y;
  }
}

// ---------------- host orchestration ----------------
extern "C" void kernel_launch(void* const* d_in, const int* in_sizes, int n_in,
                              void* d_out, int out_size, void* d_ws, size_t ws_size,
                              hipStream_t stream) {
  const float* x      = (const float*)d_in[0];
  const float* W_in   = (const float*)d_in[1];
  const float* conv_w = (const float*)d_in[2];
  const float* conv_b = (const float*)d_in[3];
  const float* W_x    = (const float*)d_in[4];
  const float* W_dt   = (const float*)d_in[5];
  const float* b_dt   = (const float*)d_in[6];
  const float* A_log  = (const float*)d_in[7];
  const float* Dp     = (const float*)d_in[8];
  const float* W_out  = (const float*)d_in[9];

  char* base = (char*)d_ws;
  size_t off = 0;
  auto alloc = [&](size_t bytes) -> void* {
    void* p = base + off;
    off += (bytes + 255) & ~(size_t)255;
    return p;
  };

  bf16_t* xn      = (bf16_t*)alloc((size_t)BL * D_SZ * 2);
  bf16_t* wbT_in  = (bf16_t*)alloc((size_t)4096 * 1024 * 2);
  bf16_t* wbT_x   = (bf16_t*)alloc((size_t)96 * 2048 * 2);
  bf16_t* wbT_dt  = (bf16_t*)alloc((size_t)2048 * 64 * 2);
  bf16_t* wbT_out = (bf16_t*)alloc((size_t)1024 * 2048 * 2);
  float*  xr      = (float*)alloc((size_t)BL * (2 * DI) * 4);
  bf16_t* ub      = (bf16_t*)alloc((size_t)BL * DI * 2);
  float*  dbl     = (float*)alloc((size_t)BL * NGRP * 4);
  bf16_t* dtb     = (bf16_t*)alloc((size_t)BL * DTR * 2);
  float*  delta   = (float*)alloc((size_t)BL * DI * 4);
  float*  ypart   = (float*)alloc((size_t)BL * DI * 4);
  bf16_t* yb      = (bf16_t*)alloc((size_t)BL * DI * 2);
  float*  hF      = (float*)alloc((size_t)B_SZ * GCH * DI * DS * 4);
  float*  Ssum    = (float*)alloc((size_t)B_SZ * GCH * DI * 4);
  float*  hstart  = (float*)alloc((size_t)B_SZ * GCH * DI * DS * 4);
  float*  xbuf0   = (float*)alloc((size_t)BL * D_SZ * 4);
  float*  xbuf1   = (float*)alloc((size_t)BL * D_SZ * 4);

  // one-time weight transpose/convert (re-done each launch; deterministic)
  k_wT<<<(1024 * 4096 + 255) / 256, 256, 0, stream>>>(W_in,  wbT_in,  1024, 4096);
  k_wT<<<(2048 * 96   + 255) / 256, 256, 0, stream>>>(W_x,   wbT_x,   2048, 96);
  k_wT<<<(64 * 2048   + 255) / 256, 256, 0, stream>>>(W_dt,  wbT_dt,  64,   2048);
  k_wT<<<(2048 * 1024 + 255) / 256, 256, 0, stream>>>(W_out, wbT_out, 2048, 1024);

  const float* xin = x;
  for (int it = 0; it < 4; ++it) {
    k_rmsnorm<<<BL, 256, 0, stream>>>(xin, xn);

    k_gemm_bf16<<<dim3(4096 / BN, BL / BM), 256, 0, stream>>>(
        xn, wbT_in, xr, BL, 4096, 1024, 1024, 4096);

    k_conv_silu<<<(BL * DI) / 256, 256, 0, stream>>>(xr, conv_w, conv_b, ub);

    k_gemm_bf16<<<dim3((96 + BN - 1) / BN, BL / BM), 256, 0, stream>>>(
        ub, wbT_x, dbl, BL, 96, 2048, 2048, 96);

    k_extract_dt<<<(BL * DTR) / 256, 256, 0, stream>>>(dbl, dtb);

    k_gemm_bf16<<<dim3(2048 / BN, BL / BM), 256, 0, stream>>>(
        dtb, wbT_dt, delta, BL, 2048, 64, 64, 2048);

    k_softplus_bias<<<(BL * DI) / 256, 256, 0, stream>>>(delta, b_dt);

    k_scan1<<<dim3(DI / 256, B_SZ, GCH), 256, 0, stream>>>(
        delta, ub, dbl, A_log, ypart, hF, Ssum);
    k_scan2<<<(B_SZ * DI) / 256, 256, 0, stream>>>(hF, Ssum, A_log, hstart);
    k_scan3<<<dim3(DI / 256, B_SZ, GCH), 256, 0, stream>>>(
        delta, ub, dbl, hstart, A_log, Dp, xr, ypart, yb);

    float* xout = (it == 3) ? (float*)d_out : ((it & 1) ? xbuf1 : xbuf0);
    k_gemm_bf16<<<dim3(1024 / BN, BL / BM), 256, 0, stream>>>(
        yb, wbT_out, xout, BL, 1024, 2048, 2048, 1024);
    xin = xout;
  }
}
